// CustomNeighborhoodAttention3D_66408784331353
// MI455X (gfx1250) — matrix operand order repaired and checked by reading, not compiled, not run on hardware
//
#include <hip/hip_runtime.h>
#include <hip/hip_bf16.h>

typedef __attribute__((ext_vector_type(16))) __bf16 v16bf;
typedef __attribute__((ext_vector_type(8)))  float  v8f;

#define DIMC   256
#define NQKV   768
#define DD     4
#define HH     32
#define WW     48
#define NTOK   (DD*HH*WW)      // 6144
#define NHEADS 4
#define HD     64
#define KDZ    3
#define KHY    5
#define KWX    5
#define NBR    (KDZ*KHY*KWX)   // 75
#define ATT_SCALE 0.125f       // 64^-0.5

// ---------------------------------------------------------------------------
// C = (A @ B + bias), columns < SCALE_COLS additionally scaled by `scale`.
// A: MxK fp32 row-major. B: KxN fp32 row-major. N,K compile-time so all
// strided accesses become immediate-offset global loads/stores and the K
// loop fully unrolls into K/32 static v_wmma_f32_16x16x32_bf16.
// Block = 256 threads = 8 waves; per-wave 16x16 C tile in a 64(M)x32(N)
// block tile.
// ---------------------------------------------------------------------------
template <int N, int K, int SCALE_COLS>
__global__ __launch_bounds__(256)
void gemm_bias_wmma_bf16(const float* __restrict__ A, const float* __restrict__ B,
                         const float* __restrict__ bias, float* __restrict__ C,
                         float scale)
{
    const int lane = threadIdx.x & 31;
    const int wave = threadIdx.x >> 5;                 // 0..7
    const int m0   = blockIdx.x * 64 + (wave >> 1) * 16;
    const int n0   = blockIdx.y * 32 + (wave & 1) * 16;
    const int half = lane >> 4;                        // 0 | 1
    const int l16  = lane & 15;

    // A fragment (16-bit A 16x32, contiguous k order chosen here, matched by B):
    //   lanes 0-15 : row=l16, K = kb+[0..15]
    //   lanes16-31 : row=l16, K = kb+[16..31]
    const float* arow = A + (size_t)(m0 + l16) * K + half * 16;
    // B fragment (16-bit B 32x16):
    //   lanes 0-15 : col=n0+l16, K = kb+[0..15]
    //   lanes16-31 : col=n0+l16, K = kb+[16..31]
    const float* bcol = B + (size_t)(half * 16) * N + (n0 + l16);

    v8f acc = {};
    #pragma unroll
    for (int kb = 0; kb < K; kb += 32) {
        const float* ap = arow + kb;
        float av[16];
        #pragma unroll
        for (int e = 0; e < 8; ++e) av[e]     = ap[e];
        #pragma unroll
        for (int e = 0; e < 8; ++e) av[8 + e] = ap[8 + e];

        const float* bp = bcol + (size_t)kb * N;
        float bv[16];
        #pragma unroll
        for (int e = 0; e < 16; ++e) bv[e] = bp[(size_t)e * N];   // imm offsets

        v16bf afrag, bfrag;
        #pragma unroll
        for (int e = 0; e < 16; ++e) {
            afrag[e] = (__bf16)av[e];
            bfrag[e] = (__bf16)bv[e];
        }
        acc = __builtin_amdgcn_wmma_f32_16x16x32_bf16(
            /*neg_a=*/false, afrag, /*neg_b=*/false, bfrag,
            /*c_mod=*/(short)0, acc, /*reuse_a=*/false, /*reuse_b=*/false);
    }

    // C/D layout (32-bit 16x16): VGPR r -> row r (lanes 0-15) / row 8+r (16-31)
    const int crow = m0 + half * 8;
    const int ccol = n0 + l16;
    const float badd = bias[ccol];
    float s = 1.0f;
    if (SCALE_COLS > 0) s = (ccol < SCALE_COLS) ? scale : 1.0f;
    float* cp = C + (size_t)crow * N + ccol;
    #pragma unroll
    for (int r = 0; r < 8; ++r) {
        cp[(size_t)r * N] = (acc[r] + badd) * s;                  // imm offsets
    }
}

// ---------------------------------------------------------------------------
// NATTEN-3D attention core. One block (128 thr) per token; wave w = head w.
// qkv laid out [token][3*256] with q|k|v each [head][64]; q pre-scaled.
// All LDS traffic is wave-private (per-head regions) -> no block barrier.
// ---------------------------------------------------------------------------
__global__ __launch_bounds__(128)
void natten3d_softmax_av(const float* __restrict__ qkv, float* __restrict__ aout)
{
    const int t    = blockIdx.x;
    const int head = threadIdx.x >> 5;
    const int lane = threadIdx.x & 31;

    const int d  = t / (HH * WW);
    const int r0 = t - d * (HH * WW);
    const int h  = r0 / WW;
    const int w  = r0 - h * WW;

    int sd = d - (KDZ / 2); sd = sd < 0 ? 0 : sd; if (sd > DD - KDZ) sd = DD - KDZ;
    int sh = h - (KHY / 2); sh = sh < 0 ? 0 : sh; if (sh > HH - KHY) sh = HH - KHY;
    int sw = w - (KWX / 2); sw = sw < 0 ? 0 : sw; if (sw > WW - KWX) sw = WW - KWX;

    __shared__ float qs[NHEADS * HD];
    __shared__ float ps[NHEADS * 80];
    __shared__ int   nts[NHEADS * 80];

    // Phase 1: this head's (pre-scaled) q into LDS
    const float* qp = qkv + (size_t)t * NQKV + head * HD;
    qs[head * HD + lane]      = qp[lane];
    qs[head * HD + lane + 32] = qp[lane + 32];

    // Phase 2: scores, lane-per-neighbor (slots j, j+32, j+64)
    float sc[3];
    #pragma unroll
    for (int s = 0; s < 3; ++s) sc[s] = -1e30f;

    const float4* qs4 = reinterpret_cast<const float4*>(&qs[head * HD]);
    #pragma unroll
    for (int s = 0; s < 3; ++s) {
        const int j = lane + s * 32;
        if (j < NBR) {
            const int dz = j / 25;
            const int rr = j - dz * 25;
            const int hy = rr / 5;
            const int wx = rr - hy * 5;
            const int nt = (sd + dz) * (HH * WW) + (sh + hy) * WW + (sw + wx);
            nts[head * 80 + j] = nt;
            const float4* kp4 = reinterpret_cast<const float4*>(
                qkv + (size_t)nt * NQKV + DIMC + head * HD);
            float a = 0.f;
            #pragma unroll
            for (int c4 = 0; c4 < HD / 4; ++c4) {
                float4 qv = qs4[c4];
                float4 kv = kp4[c4];
                a += qv.x * kv.x + qv.y * kv.y + qv.z * kv.z + qv.w * kv.w;
            }
            sc[s] = a;
        }
    }

    // Phase 3: wave32 softmax over 75 entries
    float m = fmaxf(fmaxf(sc[0], sc[1]), sc[2]);
    #pragma unroll
    for (int off = 16; off >= 1; off >>= 1)
        m = fmaxf(m, __shfl_xor(m, off, 32));

    float p[3];
    float sum = 0.f;
    #pragma unroll
    for (int s = 0; s < 3; ++s) {
        const int j = lane + s * 32;
        p[s] = (j < NBR) ? __expf(sc[s] - m) : 0.f;
        sum += p[s];
    }
    #pragma unroll
    for (int off = 16; off >= 1; off >>= 1)
        sum += __shfl_xor(sum, off, 32);
    const float inv = 1.0f / sum;
    #pragma unroll
    for (int s = 0; s < 3; ++s) {
        const int j = lane + s * 32;
        if (j < NBR) ps[head * 80 + j] = p[s] * inv;
    }

    // Phase 4: AV, lane-per-channel (c = lane, lane+32), coalesced V loads
    float acc0 = 0.f, acc1 = 0.f;
    for (int j = 0; j < NBR; ++j) {
        const float pj = ps[head * 80 + j];
        const int   nt = nts[head * 80 + j];
        const float* vp = qkv + (size_t)nt * NQKV + 2 * DIMC + head * HD;
        acc0 += pj * vp[lane];
        acc1 += pj * vp[lane + 32];
    }
    aout[(size_t)t * DIMC + head * HD + lane]      = acc0;
    aout[(size_t)t * DIMC + head * HD + lane + 32] = acc1;
}

// ---------------------------------------------------------------------------
extern "C" void kernel_launch(void* const* d_in, const int* in_sizes, int n_in,
                              void* d_out, int out_size, void* d_ws, size_t ws_size,
                              hipStream_t stream) {
    const float* x      = (const float*)d_in[0];   // 6144 x 256
    const float* w_qkv  = (const float*)d_in[1];   // 256 x 768
    const float* b_qkv  = (const float*)d_in[2];   // 768
    const float* w_proj = (const float*)d_in[3];   // 256 x 256
    const float* b_proj = (const float*)d_in[4];   // 256

    float* qkv  = (float*)d_ws;                    // 6144*768 fp32
    float* attn = qkv + (size_t)NTOK * NQKV;       // 6144*256 fp32
    float* out  = (float*)d_out;

    // 1) QKV projection (+bias, Q pre-scaled by HD^-0.5)
    gemm_bias_wmma_bf16<NQKV, DIMC, DIMC>
        <<<dim3(NTOK / 64, NQKV / 32), 256, 0, stream>>>(
            x, w_qkv, b_qkv, qkv, ATT_SCALE);

    // 2) Neighborhood attention (scores + softmax + AV)
    natten3d_softmax_av<<<NTOK, 128, 0, stream>>>(qkv, attn);

    // 3) Output projection
    gemm_bias_wmma_bf16<DIMC, DIMC, 0>
        <<<dim3(NTOK / 64, DIMC / 32), 256, 0, stream>>>(
            attn, w_proj, b_proj, out, 1.0f);
}
